// NodeCentricNewReasoner_56143812493770
// MI455X (gfx1250) — matrix operand order repaired and checked by reading, not compiled, not run on hardware
//
#include <hip/hip_runtime.h>
#include <math.h>

#define HD 128
#define NLAYERS 3
#define LN_EPS 1e-5f
#define KDEC 288   // decoder K padded: 264 -> 9 chunks of 32

typedef __attribute__((ext_vector_type(16))) __bf16 v16bf;
typedef __attribute__((ext_vector_type(8)))  float  v8f;

// ---------------- bf16 helpers (round-to-nearest-even) ----------------
static __device__ __forceinline__ unsigned short f32_to_bf16_bits(float f) {
    union { float f; unsigned u; } a; a.f = f;
    unsigned r = a.u + 0x7FFFu + ((a.u >> 16) & 1u);
    return (unsigned short)(r >> 16);
}
static __device__ __forceinline__ unsigned pack_bf16x2(float lo, float hi) {
    return (unsigned)f32_to_bf16_bits(lo) | ((unsigned)f32_to_bf16_bits(hi) << 16);
}
static __device__ __forceinline__ float bf16_bits_to_f32(unsigned short s) {
    return __uint_as_float((unsigned)s << 16);
}

// A fragment from a ROW-MAJOR bf16 row (16-bit A 16x32, contiguous K order):
// lane<16 : elems 0-15 = K k0..k0+15
// lane>=16: elems 0-15 = K k0+16..k0+31.  Two aligned 16-byte loads, no conversion.
static __device__ __forceinline__ v16bf load_a_frag(const unsigned short* rowPtr,
                                                    int k0, int laneHi) {
    const unsigned short* p = rowPtr + k0 + (laneHi ? 16 : 0);
    union { v16bf v; struct { uint4 lo; uint4 hi; } u; } f;
    f.u.lo = *(const uint4*)(p);
    f.u.hi = *(const uint4*)(p + 8);
    return f.v;
}

// B fragment from transposed bf16 weights W_T[n][k] (ldk multiple of 16):
// lane<16 holds K k0..k0+15 of column n, lane>=16 holds k0+16..k0+31.
static __device__ __forceinline__ v16bf load_b_frag(const unsigned short* WT, int ldk,
                                                    int n, int k0, int laneHi) {
    const unsigned short* p = WT + (size_t)n * ldk + k0 + (laneHi ? 16 : 0);
    return *(const v16bf*)p;
}

static __device__ __forceinline__ v8f wmma_bf16(v16bf a, v16bf b, v8f c) {
    return __builtin_amdgcn_wmma_f32_16x16x32_bf16(false, a, false, b, (short)0, c, false, false);
}

// ---------------- generic f32 -> bf16 convert (8 elems/thread) ----------------
__global__ void cvt_bf16_kernel(const float* __restrict__ in,
                                unsigned short* __restrict__ out, long n) {
    long i = ((long)blockIdx.x * blockDim.x + threadIdx.x) * 8;
    long stride = (long)gridDim.x * blockDim.x * 8;
    for (long j = i; j < n; j += stride) {
        float4 a = *(const float4*)(in + j);
        float4 b = *(const float4*)(in + j + 4);
        uint4 o;
        o.x = pack_bf16x2(a.x, a.y);
        o.y = pack_bf16x2(a.z, a.w);
        o.z = pack_bf16x2(b.x, b.y);
        o.w = pack_bf16x2(b.z, b.w);
        *(uint4*)(out + j) = o;
    }
}

// ---------------- weight prep ----------------
__global__ void prep_c0_kernel(const float* __restrict__ Wf, const float* __restrict__ bf,
                               float* __restrict__ c0) {
    int j = threadIdx.x;                       // 128 threads
    float s = bf[j];
    for (int i = 0; i < HD; ++i) s += Wf[i * HD + j];   // ones @ Wf[:128]
    c0[j] = s;
}

__global__ void prep_WfT_kernel(const float* __restrict__ Wf, unsigned short* __restrict__ WfT) {
    int idx = blockIdx.x * blockDim.x + threadIdx.x;    // 128*128
    int n = idx >> 7, k = idx & 127;
    WfT[n * HD + k] = f32_to_bf16_bits(Wf[(HD + k) * HD + n]);
}

__global__ void prep_WcatT_kernel(const float* __restrict__ Wrel, const float* __restrict__ Wroot,
                                  unsigned short* __restrict__ WcatT) {
    int idx = blockIdx.x * blockDim.x + threadIdx.x;    // 3*128*256
    int l = idx / (HD * 2 * HD);
    int rem = idx % (HD * 2 * HD);
    int n = rem / (2 * HD);
    int k = rem % (2 * HD);
    float v = (k < HD) ? Wrel[((size_t)l * HD + k) * HD + n]
                       : Wroot[((size_t)l * HD + (k - HD)) * HD + n];
    WcatT[((size_t)l * HD + n) * (2 * HD) + k] = f32_to_bf16_bits(v);
}

__global__ void prep_Wd1T_kernel(const float* __restrict__ Wd1, unsigned short* __restrict__ Wd1T) {
    int idx = blockIdx.x * blockDim.x + threadIdx.x;    // 128*288
    int n = idx / KDEC, k = idx % KDEC;
    float v = (k < 2 * HD + 8) ? Wd1[(size_t)k * HD + n] : 0.0f;
    Wd1T[(size_t)n * KDEC + k] = f32_to_bf16_bits(v);
}

// ---------------- per-edge weight MLP (tiny, VALU) ----------------
__global__ __launch_bounds__(256) void edge_weight_kernel(
        const float* __restrict__ af, const float* __restrict__ We1,
        const float* __restrict__ be1, const float* __restrict__ We2,
        const float* __restrict__ be2, float* __restrict__ ew, int E) {
    __shared__ float sW[8 * HD];
    __shared__ float sb1[HD];
    __shared__ float sW2[HD];
    for (int i = threadIdx.x; i < 8 * HD; i += blockDim.x) sW[i] = We1[i];
    for (int i = threadIdx.x; i < HD; i += blockDim.x) { sb1[i] = be1[i]; sW2[i] = We2[i]; }
    __syncthreads();
    int e = blockIdx.x * blockDim.x + threadIdx.x;
    if (e >= E) return;
    const float4* p = (const float4*)(af + (size_t)e * 8);
    float4 q0 = p[0], q1 = p[1];
    float a[8] = { q0.x, q0.y, q0.z, q0.w, q1.x, q1.y, q1.z, q1.w };
    float acc = 0.0f;
    #pragma unroll 4
    for (int j = 0; j < HD; ++j) {
        float h = sb1[j];
        #pragma unroll
        for (int k = 0; k < 8; ++k) h = fmaf(a[k], sW[k * HD + j], h);
        h = fmaxf(h, 0.0f);
        acc = fmaf(h, sW2[j], acc);
    }
    float z = acc + be2[0];
    ew[e] = 1.0f / (1.0f + expf(-z));
}

// ---------------- node fusion: x = relu(h_old @ WfT + c0) ----------------
__global__ __launch_bounds__(256) void fusion_gemm_kernel(
        const unsigned short* __restrict__ holdh, const unsigned short* __restrict__ WfT,
        const float* __restrict__ c0, float* __restrict__ x,
        unsigned short* __restrict__ xh, int M) {
    int lane = threadIdx.x & 31, wv = threadIdx.x >> 5;
    int laneHi = lane >> 4, lm = lane & 15;
    int rowBase = blockIdx.x * 16;
    int r = rowBase + lm; if (r >= M) r = M - 1;
    const unsigned short* rowPtr = holdh + (size_t)r * HD;
    int n = wv * 16 + lm;
    v8f acc = {};
    #pragma unroll
    for (int c = 0; c < 4; ++c) {
        v16bf a = load_a_frag(rowPtr, c * 32, laneHi);
        v16bf b = load_b_frag(WfT, HD, n, c * 32, laneHi);
        acc = wmma_bf16(a, b, acc);
    }
    float bias = c0[n];
    #pragma unroll
    for (int v = 0; v < 8; ++v) {
        int row = rowBase + v + 8 * laneHi;
        if (row < M) {
            float y = fmaxf(acc[v] + bias, 0.0f);
            x[(size_t)row * HD + n]  = y;
            xh[(size_t)row * HD + n] = f32_to_bf16_bits(y);
        }
    }
}

// ---------------- zero scratch ----------------
__global__ void zero_kernel(float* __restrict__ p, long n) {
    long i = ((long)blockIdx.x * blockDim.x + threadIdx.x) * 4;
    long stride = (long)gridDim.x * blockDim.x * 4;
    for (long j = i; j < n; j += stride)
        *(float4*)(p + j) = make_float4(0.f, 0.f, 0.f, 0.f);
}

// ---------------- message scatter: agg[dst] += ew * x[src]  (bf16 gather) ----------------
__global__ __launch_bounds__(256) void scatter_kernel(
        const int* __restrict__ src, const int* __restrict__ dst,
        const float* __restrict__ ew, const unsigned short* __restrict__ xh,
        float* __restrict__ agg, int E) {
    int e = (blockIdx.x * blockDim.x + threadIdx.x) >> 5;
    int lane = threadIdx.x & 31;
    if (e >= E) return;
    int s = src[e], d = dst[e];
    float w = ew[e];
    ushort4 v = *(const ushort4*)(xh + (size_t)s * HD + lane * 4);
    float* o = agg + (size_t)d * HD + lane * 4;
    atomicAdd(o + 0, w * bf16_bits_to_f32(v.x));
    atomicAdd(o + 1, w * bf16_bits_to_f32(v.y));
    atomicAdd(o + 2, w * bf16_bits_to_f32(v.z));
    atomicAdd(o + 3, w * bf16_bits_to_f32(v.w));
}

// ---------------- layer: h = [agg|x] @ Wcat + brel ; x' = relu(LN(h)) + x ----------------
__global__ __launch_bounds__(256) void layer_gemm_kernel(
        const unsigned short* __restrict__ aggh, const unsigned short* __restrict__ xhin,
        const float* __restrict__ xin, const unsigned short* __restrict__ WcatT,
        const float* __restrict__ brel, const float* __restrict__ gamma,
        const float* __restrict__ beta, float* __restrict__ xout,
        unsigned short* __restrict__ xhout, int M) {
    __shared__ float tile[16][HD + 4];
    int lane = threadIdx.x & 31, wv = threadIdx.x >> 5;
    int laneHi = lane >> 4, lm = lane & 15;
    int rowBase = blockIdx.x * 16;
    int r = rowBase + lm; if (r >= M) r = M - 1;
    const unsigned short* aggRow = aggh + (size_t)r * HD;
    const unsigned short* xRow   = xhin + (size_t)r * HD;
    int n = wv * 16 + lm;
    v8f acc = {};
    #pragma unroll
    for (int c = 0; c < 8; ++c) {
        const unsigned short* rp = (c < 4) ? aggRow : xRow;
        v16bf a = load_a_frag(rp, (c & 3) * 32, laneHi);
        v16bf b = load_b_frag(WcatT, 2 * HD, n, c * 32, laneHi);
        acc = wmma_bf16(a, b, acc);
    }
    float bias = brel[n];
    #pragma unroll
    for (int v = 0; v < 8; ++v) tile[v + 8 * laneHi][n] = acc[v] + bias;
    __syncthreads();
    // LayerNorm + relu + residual: wave wv handles rows 2wv, 2wv+1; 4 elems/lane
    for (int rr = 2 * wv; rr < 2 * wv + 2; ++rr) {
        float vals[4];
        #pragma unroll
        for (int i = 0; i < 4; ++i) vals[i] = tile[rr][lane * 4 + i];
        float s  = vals[0] + vals[1] + vals[2] + vals[3];
        float s2 = vals[0]*vals[0] + vals[1]*vals[1] + vals[2]*vals[2] + vals[3]*vals[3];
        #pragma unroll
        for (int off = 1; off < 32; off <<= 1) {
            s  += __shfl_xor(s,  off);
            s2 += __shfl_xor(s2, off);
        }
        float mu  = s * (1.0f / HD);
        float var = s2 * (1.0f / HD) - mu * mu;
        float inv = rsqrtf(var + LN_EPS);
        int row = rowBase + rr;
        if (row < M) {
            int cidx = lane * 4;
            float y0 = fmaxf((vals[0] - mu) * inv * gamma[cidx + 0] + beta[cidx + 0], 0.0f)
                       + xin[(size_t)row * HD + cidx + 0];
            float y1 = fmaxf((vals[1] - mu) * inv * gamma[cidx + 1] + beta[cidx + 1], 0.0f)
                       + xin[(size_t)row * HD + cidx + 1];
            float y2 = fmaxf((vals[2] - mu) * inv * gamma[cidx + 2] + beta[cidx + 2], 0.0f)
                       + xin[(size_t)row * HD + cidx + 2];
            float y3 = fmaxf((vals[3] - mu) * inv * gamma[cidx + 3] + beta[cidx + 3], 0.0f)
                       + xin[(size_t)row * HD + cidx + 3];
            *(float4*)(xout + (size_t)row * HD + cidx) = make_float4(y0, y1, y2, y3);
            uint2 hpack;
            hpack.x = pack_bf16x2(y0, y1);
            hpack.y = pack_bf16x2(y2, y3);
            *(uint2*)(xhout + (size_t)row * HD + cidx) = hpack;
        }
    }
}

// ---------------- decoder: out = relu([x_s|x_d|af] @ Wd1 + bd1) @ Wd2 + bd2 ----------------
__global__ __launch_bounds__(256) void decoder_kernel(
        const unsigned short* __restrict__ xh, const unsigned short* __restrict__ afh,
        const int* __restrict__ src, const int* __restrict__ dst,
        const unsigned short* __restrict__ Wd1T, const float* __restrict__ bd1,
        const float* __restrict__ Wd2, const float* __restrict__ bd2,
        float* __restrict__ out, int E) {
    __shared__ float red[16][8];     // [row][wave]
    int lane = threadIdx.x & 31, wv = threadIdx.x >> 5;
    int laneHi = lane >> 4, lm = lane & 15;
    long eb = (long)blockIdx.x * 16;
    long et = eb + lm; if (et > (long)E - 1) et = (long)E - 1;
    int e = (int)et;
    const unsigned short* xs = xh + (size_t)src[e] * HD;
    const unsigned short* xd = xh + (size_t)dst[e] * HD;
    int n = wv * 16 + lm;
    v8f acc = {};
    #pragma unroll
    for (int c = 0; c < 8; ++c) {
        const unsigned short* rp = (c < 4) ? xs : xd;
        v16bf a = load_a_frag(rp, (c & 3) * 32, laneHi);
        v16bf b = load_b_frag(Wd1T, KDEC, n, c * 32, laneHi);
        acc = wmma_bf16(a, b, acc);
    }
    {   // K chunk 8: global K 256..287 = [af(8) | zero pad]
        union { v16bf v; struct { uint4 lo; uint4 hi; } u; } f;
        f.u.lo = make_uint4(0u, 0u, 0u, 0u);
        f.u.hi = make_uint4(0u, 0u, 0u, 0u);
        if (!laneHi) f.u.lo = *(const uint4*)(afh + (size_t)e * 8);
        v16bf b = load_b_frag(Wd1T, KDEC, n, 256, laneHi);
        acc = wmma_bf16(f.v, b, acc);
    }
    float bias = bd1[n], w2 = Wd2[n];
    float part[8];
    #pragma unroll
    for (int v = 0; v < 8; ++v) part[v] = fmaxf(acc[v] + bias, 0.0f) * w2;
    // reduce across the 16 lanes of each half-wave (xor masks < 16 stay in-group)
    #pragma unroll
    for (int v = 0; v < 8; ++v) {
        float t = part[v];
        t += __shfl_xor(t, 1); t += __shfl_xor(t, 2);
        t += __shfl_xor(t, 4); t += __shfl_xor(t, 8);
        part[v] = t;
    }
    if (lm == 0) {
        #pragma unroll
        for (int v = 0; v < 8; ++v) red[v + 8 * laneHi][wv] = part[v];
    }
    __syncthreads();
    if (threadIdx.x < 16) {
        long ee = eb + threadIdx.x;
        if (ee < E) {
            float s = 0.0f;
            #pragma unroll
            for (int w = 0; w < 8; ++w) s += red[threadIdx.x][w];
            out[ee] = s + bd2[0];
        }
    }
}

// ---------------- launch ----------------
extern "C" void kernel_launch(void* const* d_in, const int* in_sizes, int n_in,
                              void* d_out, int out_size, void* d_ws, size_t ws_size,
                              hipStream_t stream) {
    (void)n_in; (void)out_size; (void)ws_size;
    const int*   ei    = (const int*)  d_in[0];
    const float* af    = (const float*)d_in[1];
    const float* hold  = (const float*)d_in[2];
    const float* Wf    = (const float*)d_in[4];
    const float* bfv   = (const float*)d_in[5];
    const float* We1   = (const float*)d_in[6];
    const float* be1   = (const float*)d_in[7];
    const float* We2   = (const float*)d_in[8];
    const float* be2   = (const float*)d_in[9];
    const float* Wrel  = (const float*)d_in[10];
    const float* brel  = (const float*)d_in[11];
    const float* Wroot = (const float*)d_in[12];
    const float* gamma = (const float*)d_in[13];
    const float* beta  = (const float*)d_in[14];
    const float* Wd1   = (const float*)d_in[15];
    const float* bd1   = (const float*)d_in[16];
    const float* Wd2   = (const float*)d_in[17];
    const float* bd2   = (const float*)d_in[18];

    const int N = in_sizes[2] / HD;
    const int E = in_sizes[1] / 8;
    const int* src = ei;
    const int* dst = ei + E;

    char* ws = (char*)d_ws;
    size_t off = 0;
    auto alloc = [&](size_t bytes) -> void* {
        void* p = ws + off;
        off = (off + bytes + 255) & ~(size_t)255;
        return p;
    };
    float*          x0    = (float*)alloc((size_t)N * HD * 4);
    float*          x1    = (float*)alloc((size_t)N * HD * 4);
    float*          agg   = (float*)alloc((size_t)N * HD * 4);
    float*          ew    = (float*)alloc((size_t)E * 4);
    unsigned short* xh0   = (unsigned short*)alloc((size_t)N * HD * 2);
    unsigned short* xh1   = (unsigned short*)alloc((size_t)N * HD * 2);
    unsigned short* aggh  = (unsigned short*)alloc((size_t)N * HD * 2);
    unsigned short* holdh = (unsigned short*)alloc((size_t)N * HD * 2);
    unsigned short* afh   = (unsigned short*)alloc((size_t)E * 8 * 2);
    float*          c0    = (float*)alloc(HD * 4);
    unsigned short* WfT   = (unsigned short*)alloc((size_t)HD * HD * 2);
    unsigned short* WcatT = (unsigned short*)alloc((size_t)NLAYERS * HD * 2 * HD * 2);
    unsigned short* Wd1T  = (unsigned short*)alloc((size_t)HD * KDEC * 2);

    const long nNode = (long)N * HD;   // multiple of 8
    const long nAf   = (long)E * 8;    // multiple of 8
    const int  cvtBlkNode = (int)((nNode / 8 + 255) / 256);
    const int  cvtBlkAf   = (int)((nAf / 8 + 255) / 256);

    // weight prep (bf16 transposed layouts for WMMA B fragments)
    prep_c0_kernel<<<1, 128, 0, stream>>>(Wf, bfv, c0);
    prep_WfT_kernel<<<(HD * HD) / 256, 256, 0, stream>>>(Wf, WfT);
    prep_WcatT_kernel<<<(NLAYERS * HD * 2 * HD) / 256, 256, 0, stream>>>(Wrel, Wroot, WcatT);
    prep_Wd1T_kernel<<<(HD * KDEC) / 256, 256, 0, stream>>>(Wd1, Wd1T);

    // one-time bf16 copies of activations/features
    cvt_bf16_kernel<<<cvtBlkNode, 256, 0, stream>>>(hold, holdh, nNode);
    cvt_bf16_kernel<<<cvtBlkAf, 256, 0, stream>>>(af, afh, nAf);

    // edge weights + node fusion
    edge_weight_kernel<<<(E + 255) / 256, 256, 0, stream>>>(af, We1, be1, We2, be2, ew, E);
    fusion_gemm_kernel<<<(N + 15) / 16, 256, 0, stream>>>(holdh, WfT, c0, x0, xh0, N);

    float*          xc  = x0;
    float*          xn  = x1;
    unsigned short* xhc = xh0;
    unsigned short* xhn = xh1;
    for (int l = 0; l < NLAYERS; ++l) {
        zero_kernel<<<(int)((nNode / 4 + 255) / 256), 256, 0, stream>>>(agg, nNode);
        scatter_kernel<<<(E + 7) / 8, 256, 0, stream>>>(src, dst, ew, xhc, agg, E);
        cvt_bf16_kernel<<<cvtBlkNode, 256, 0, stream>>>(agg, aggh, nNode);
        layer_gemm_kernel<<<(N + 15) / 16, 256, 0, stream>>>(
            aggh, xhc, xc, WcatT + (size_t)l * HD * 2 * HD,
            brel + (size_t)l * HD, gamma + (size_t)l * HD, beta + (size_t)l * HD,
            xn, xhn, N);
        { float* t = xc; xc = xn; xn = t; }
        { unsigned short* t = xhc; xhc = xhn; xhn = t; }
    }

    decoder_kernel<<<(E + 15) / 16, 256, 0, stream>>>(
        xhc, afh, src, dst, Wd1T, bd1, Wd2, bd2, (float*)d_out, E);
}
